// TransducerBeamSearcher_68607807587019
// MI455X (gfx1250) — matrix operand order repaired and checked by reading, compiled, not run on hardware
//
#include <hip/hip_runtime.h>
#include <hip/hip_bf16.h>
#include <math.h>

// MI455X / gfx1250, wave32. All matrix math via v_wmma_f32_16x16x32_f16.
// B=32, T=1000, E=H=J=512, V=1024.

typedef __attribute__((ext_vector_type(16))) _Float16 v16h;
typedef __attribute__((ext_vector_type(8)))  float    v8f;

union FragU {
  v16h h;
  float4 f4[2];
};

__device__ __forceinline__ v8f wmma_f16(v16h a, v16h b, v8f c) {
  return __builtin_amdgcn_wmma_f32_16x16x32_f16(false, a, false, b, (short)0, c,
                                                false, false);
}

// ---- gfx1250 async global->LDS copy (ASYNCcnt path). Signature probed:
// ---- (v4i AS1*, v4i AS3*, imm offset, imm cpol), GCC-style vector_size(16).
#if defined(__HIP_DEVICE_COMPILE__) && \
    __has_builtin(__builtin_amdgcn_global_load_async_to_lds_b128)
#define HAVE_ASYNC_LDS 1
typedef int cdna5_v4i __attribute__((vector_size(16)));
typedef cdna5_v4i __attribute__((address_space(1))) * cdna5_gptr_v4i;
typedef cdna5_v4i __attribute__((address_space(3))) * cdna5_lptr_v4i;
__device__ __forceinline__ void async_cp16(void* dst_lds, const void* src_g) {
  __builtin_amdgcn_global_load_async_to_lds_b128(
      (cdna5_gptr_v4i)src_g, (cdna5_lptr_v4i)dst_lds, 0, 0);
}
__device__ __forceinline__ void async_wait0() {
#if __has_builtin(__builtin_amdgcn_s_wait_asynccnt)
  __builtin_amdgcn_s_wait_asynccnt(0);
#else
  asm volatile("s_wait_asynccnt 0x0" ::: "memory");
#endif
}
#else
#define HAVE_ASYNC_LDS 0
#endif

// ---------------------------------------------------------------------------
// f32 -> f16 elementwise convert (encoder activations, embedding table)
// ---------------------------------------------------------------------------
__global__ void k_f32_to_f16(const float* __restrict__ src,
                             _Float16* __restrict__ dst, int n) {
  int i = blockIdx.x * blockDim.x + threadIdx.x;
  int stride = gridDim.x * blockDim.x;
  for (; i < n; i += stride) dst[i] = (_Float16)src[i];
}

// ---------------------------------------------------------------------------
// Pack a (K x N) row-major f32 weight matrix (optionally two stacked sources,
// for [W_ih;W_hh]) into per-lane WMMA B fragments, f16.
// Tile = 32(K) x 16(N). Lane L holds column n = nt*16 + (L&15), element i at
// k = kt*32 + (L>>4)*8 + (i&7) + (i>>3)*16. tileIdx = ntG*Ktiles + kt.
// ---------------------------------------------------------------------------
__global__ void k_pack_b(const float* __restrict__ srcA,
                         const float* __restrict__ srcB, int kRowsA, int K,
                         int N, float4* __restrict__ dst) {
  int gw = (blockIdx.x * blockDim.x + threadIdx.x) >> 5;
  int lane = threadIdx.x & 31;
  int Kt = K >> 5;
  int nTiles = Kt * (N >> 4);
  if (gw >= nTiles) return;
  int ntG = gw / Kt, kt = gw - ntG * Kt;
  int n = ntG * 16 + (lane & 15);
  int kBase = kt * 32 + (lane >> 4) * 8;
  FragU f;
#pragma unroll
  for (int i = 0; i < 16; ++i) {
    int k = kBase + (i & 7) + (i >> 3) * 16;
    const float* s = (k < kRowsA) ? (srcA + (size_t)k * N + n)
                                  : (srcB + (size_t)(k - kRowsA) * N + n);
    f.h[i] = (_Float16)(*s);
  }
  dst[(size_t)gw * 64 + lane * 2] = f.f4[0];
  dst[(size_t)gw * 64 + lane * 2 + 1] = f.f4[1];
}

// ---------------------------------------------------------------------------
// enc_proj = encoder(32000 x 512, f16) @ W_enc(512 x 512, packed)
// 4 N-tiles per wave; all 4 B fragments loaded into distinct registers before
// the WMMAs so the scheduler can use partial s_wait_loadcnt.
// ---------------------------------------------------------------------------
__global__ void k_enc_proj(const _Float16* __restrict__ A,
                           const float4* __restrict__ Bp,
                           float* __restrict__ encP) {
  int gw = (blockIdx.x * blockDim.x + threadIdx.x) >> 5;
  int lane = threadIdx.x & 31;
  if (gw >= 2000 * 8) return;
  int mt = gw >> 3, ng = gw & 7;  // ng selects a group of 4 n-tiles
  int ln = lane & 15, half = lane >> 4;
  v8f acc[4] = {};
  const float4* arow = (const float4*)(A + (size_t)(mt * 16 + ln) * 512);
  for (int kt = 0; kt < 16; ++kt) {
    FragU a, bf[4];
    a.f4[0] = arow[(kt * 32 + half * 8) >> 3];
    a.f4[1] = arow[(kt * 32 + 16 + half * 8) >> 3];
#pragma unroll
    for (int l = 0; l < 4; ++l) {
      size_t tIdx = (size_t)(ng * 4 + l) * 16 + kt;
      bf[l].f4[0] = Bp[tIdx * 64 + lane * 2];
      bf[l].f4[1] = Bp[tIdx * 64 + lane * 2 + 1];
    }
#pragma unroll
    for (int l = 0; l < 4; ++l) acc[l] = wmma_f16(a.h, bf[l].h, acc[l]);
  }
#pragma unroll
  for (int l = 0; l < 4; ++l) {
    int n = (ng * 4 + l) * 16 + ln;
#pragma unroll
    for (int r = 0; r < 8; ++r) {
      int m = mt * 16 + half * 8 + r;  // global row = b*1000 + t
      int bb = m / 1000, tt = m - bb * 1000;
      encP[((size_t)tt * 32 + bb) * 512 + n] = acc[l][r];
    }
  }
}

// ---------------------------------------------------------------------------
// Persistent greedy transducer decode: one workgroup, 512 threads = 16 waves,
// all state in LDS, weights streamed from L2 as packed WMMA fragments.
// Note out_PN == h at all times (identical init + identical update), so one
// f16 state buffer serves both.
// ---------------------------------------------------------------------------
__global__ __launch_bounds__(512, 1) void k_decode(
    const float* __restrict__ encP,        // (1000, 32, 512) f32
    const _Float16* __restrict__ embF16,   // (1024, 512) f16
    const float* __restrict__ b_lstm,      // (2048)
    const float* __restrict__ b_joint,     // (512)
    const float4* __restrict__ Wdec,       // packed K=512,  N=512
    const float4* __restrict__ Wout,       // packed K=512,  N=1024
    const float4* __restrict__ Wlstm,      // packed K=1024, N=2048 ([W_ih;W_hh])
    float* __restrict__ out) {             // 32*1000 preds | 32 scores | 1 mean
  __shared__ _Float16 sA[32][1024];        // [ x/joint (0..511) | h (512..1023) ]
  __shared__ float sC[32][512];            // LSTM cell state, f32
  __shared__ float sRedMax[16][32];
  __shared__ int sRedArg[16][32];
  __shared__ float sRedSum[16][32];
  __shared__ float sGMax[32];
  __shared__ int sPos[32];
  __shared__ int sEmit[32];
  __shared__ float sScore[32];

  const int tid = threadIdx.x;
  const int lane = tid & 31, w = tid >> 5;
  const int ln = lane & 15, half = lane >> 4;

  // ---- init: x = embed[BLANK], h = 0, c = 0, emit forced true ----
  for (int i = tid; i < 32 * 512; i += 512) {
    int b = i >> 9, n = i & 511;
    sC[b][n] = 0.f;
    sA[b][512 + n] = (_Float16)0.f;
    sA[b][n] = embF16[n];  // row 0 == BLANK
  }
  if (tid < 32) { sEmit[tid] = 1; sScore[tid] = 0.f; sPos[tid] = 0; }
  __syncthreads();

  float hnR[2][2][8];  // staged h_new (applied after barrier: RAW on sA h-region)

  // LSTM: gates = [x,h] @ Wlstm + b; wave w owns columns j in [w*32, w*32+32)
  // of each of the i/f/g/o blocks. All 8 B fragments of a k-step are loaded
  // into distinct registers before the 8 WMMAs (partial loadcnt waits).
  auto lstm_gemm = [&]() {
#pragma unroll
    for (int mt = 0; mt < 2; ++mt) {
      v8f acc[4][2] = {};
      const float4* arow = (const float4*)&sA[mt * 16 + ln][0];
      for (int kk = 0; kk < 32; ++kk) {
        FragU a, bf[8];
        a.f4[0] = arow[(kk * 32 + half * 8) >> 3];
        a.f4[1] = arow[(kk * 32 + 16 + half * 8) >> 3];
#pragma unroll
        for (int i = 0; i < 8; ++i) {
          int g = i >> 1, l = i & 1;
          size_t tIdx = (size_t)(g * 32 + w * 2 + l) * 32 + kk;
          bf[i].f4[0] = Wlstm[tIdx * 64 + lane * 2];
          bf[i].f4[1] = Wlstm[tIdx * 64 + lane * 2 + 1];
        }
#pragma unroll
        for (int i = 0; i < 8; ++i)
          acc[i >> 1][i & 1] = wmma_f16(a.h, bf[i].h, acc[i >> 1][i & 1]);
      }
#pragma unroll
      for (int l = 0; l < 2; ++l) {
        int j = (w * 2 + l) * 16 + ln;
        float bi = b_lstm[j], bf_ = b_lstm[512 + j];
        float bg = b_lstm[1024 + j], bo = b_lstm[1536 + j];
#pragma unroll
        for (int r = 0; r < 8; ++r) {
          int m = mt * 16 + half * 8 + r;
          float si = 1.f / (1.f + __expf(-(acc[0][l][r] + bi)));
          float sf = 1.f / (1.f + __expf(-(acc[1][l][r] + bf_)));
          float gv = tanhf(acc[2][l][r] + bg);
          float so = 1.f / (1.f + __expf(-(acc[3][l][r] + bo)));
          float cn = sf * sC[m][j] + si * gv;  // sC[m][j] owned by this wave
          float hn = so * tanhf(cn);
          if (sEmit[m]) sC[m][j] = cn;
          hnR[mt][l][r] = hn;
        }
      }
    }
  };
  auto lstm_apply = [&]() {  // after barrier: conditionally commit h (== out_PN)
#pragma unroll
    for (int mt = 0; mt < 2; ++mt)
#pragma unroll
      for (int l = 0; l < 2; ++l) {
        int j = (w * 2 + l) * 16 + ln;
#pragma unroll
        for (int r = 0; r < 8; ++r) {
          int m = mt * 16 + half * 8 + r;
          if (sEmit[m]) sA[m][512 + j] = (_Float16)hnR[mt][l][r];
        }
      }
  };

  lstm_gemm();  // initial prediction-network step (tok0 = BLANK)
  __syncthreads();
  lstm_apply();
  __syncthreads();

  for (int t = 0; t < 1000; ++t) {
    // ---- joint = tanh(enc_t + h @ W_dec + b_joint) -> sA[:,0:511] (f16) ----
    {
      v8f accd[2][2] = {};
      const float4* r0 = (const float4*)&sA[ln][512];
      const float4* r1 = (const float4*)&sA[16 + ln][512];
      for (int kt = 0; kt < 16; ++kt) {
        FragU a0, a1, bf[2];
        a0.f4[0] = r0[(kt * 32 + half * 8) >> 3];
        a0.f4[1] = r0[(kt * 32 + 16 + half * 8) >> 3];
        a1.f4[0] = r1[(kt * 32 + half * 8) >> 3];
        a1.f4[1] = r1[(kt * 32 + 16 + half * 8) >> 3];
#pragma unroll
        for (int l = 0; l < 2; ++l) {
          size_t tIdx = (size_t)(w * 2 + l) * 16 + kt;
          bf[l].f4[0] = Wdec[tIdx * 64 + lane * 2];
          bf[l].f4[1] = Wdec[tIdx * 64 + lane * 2 + 1];
        }
#pragma unroll
        for (int l = 0; l < 2; ++l) {
          accd[0][l] = wmma_f16(a0.h, bf[l].h, accd[0][l]);
          accd[1][l] = wmma_f16(a1.h, bf[l].h, accd[1][l]);
        }
      }
      const float* encT = encP + (size_t)t * 32 * 512;
#pragma unroll
      for (int mt = 0; mt < 2; ++mt)
#pragma unroll
        for (int l = 0; l < 2; ++l) {
          int n = (w * 2 + l) * 16 + ln;
          float bj = b_joint[n];
#pragma unroll
          for (int r = 0; r < 8; ++r) {
            int m = mt * 16 + half * 8 + r;
            float v = accd[mt][l][r] + encT[(size_t)m * 512 + n] + bj;
            sA[m][n] = (_Float16)tanhf(v);
          }
        }
      if (t + 1 < 1000) {  // gfx1250 global_prefetch_b8: next enc slice,
        const float* pf = encT + 32 * 512 + (size_t)tid * 32;  // 128B/thread
        __builtin_prefetch(pf, 0, 1);
        __builtin_prefetch(pf + 16, 0, 1);
      }
    }
    __syncthreads();

    // ---- logits = joint @ W_out (kept in registers) ----
    v8f accL[2][4] = {};
    {
      const float4* r0 = (const float4*)&sA[ln][0];
      const float4* r1 = (const float4*)&sA[16 + ln][0];
      for (int kt = 0; kt < 16; ++kt) {
        FragU a0, a1, bf[4];
        a0.f4[0] = r0[(kt * 32 + half * 8) >> 3];
        a0.f4[1] = r0[(kt * 32 + 16 + half * 8) >> 3];
        a1.f4[0] = r1[(kt * 32 + half * 8) >> 3];
        a1.f4[1] = r1[(kt * 32 + 16 + half * 8) >> 3];
#pragma unroll
        for (int l = 0; l < 4; ++l) {
          size_t tIdx = (size_t)(w * 4 + l) * 16 + kt;
          bf[l].f4[0] = Wout[tIdx * 64 + lane * 2];
          bf[l].f4[1] = Wout[tIdx * 64 + lane * 2 + 1];
        }
#pragma unroll
        for (int l = 0; l < 4; ++l) {
          accL[0][l] = wmma_f16(a0.h, bf[l].h, accL[0][l]);
          accL[1][l] = wmma_f16(a1.h, bf[l].h, accL[1][l]);
        }
      }
    }
    // ---- pass 1: per-row max + argmax (shuffle within 16-lane halves) ----
#pragma unroll
    for (int mt = 0; mt < 2; ++mt)
#pragma unroll
      for (int r = 0; r < 8; ++r) {
        float vmax = accL[mt][0][r];
        int varg = (w * 4) * 16 + ln;
#pragma unroll
        for (int l = 1; l < 4; ++l) {
          float v = accL[mt][l][r];
          int a = (w * 4 + l) * 16 + ln;
          if (v > vmax || (v == vmax && a < varg)) { vmax = v; varg = a; }
        }
        for (int s = 1; s < 16; s <<= 1) {
          float ov = __shfl_xor(vmax, s, 32);
          int oa = __shfl_xor(varg, s, 32);
          if (ov > vmax || (ov == vmax && oa < varg)) { vmax = ov; varg = oa; }
        }
        int m = mt * 16 + half * 8 + r;
        if (ln == 0) { sRedMax[w][m] = vmax; sRedArg[w][m] = varg; }
      }
    __syncthreads();
    if (tid < 32) {
      float g = sRedMax[0][tid];
      int a = sRedArg[0][tid];
      for (int w2 = 1; w2 < 16; ++w2) {
        float v = sRedMax[w2][tid];
        int aa = sRedArg[w2][tid];
        if (v > g || (v == g && aa < a)) { g = v; a = aa; }
      }
      sGMax[tid] = g; sPos[tid] = a; sEmit[tid] = (a != 0);
    }
    __syncthreads();
    // ---- pass 2: sum exp(l - max)  ->  lp = -log(sum) ----
#pragma unroll
    for (int mt = 0; mt < 2; ++mt)
#pragma unroll
      for (int r = 0; r < 8; ++r) {
        int m = mt * 16 + half * 8 + r;
        float g = sGMax[m];
        float s = 0.f;
#pragma unroll
        for (int l = 0; l < 4; ++l) s += __expf(accL[mt][l][r] - g);
        for (int d = 1; d < 16; d <<= 1) s += __shfl_xor(s, d, 32);
        if (ln == 0) sRedSum[w][m] = s;
      }
    __syncthreads();
    // ---- x = embed[pos]: async global->LDS copy of f16 rows (32 KB) ----
#pragma unroll
    for (int k = 0; k < 4; ++k) {
      int chunk = tid + k * 512;         // 2048 x 16B chunks
      int b = chunk >> 6;                // 64 chunks per row
      int off16 = (chunk & 63) * 8;      // offset in halves
      const _Float16* src = embF16 + (size_t)sPos[b] * 512 + off16;
      _Float16* dst = &sA[b][off16];
#if HAVE_ASYNC_LDS
      async_cp16(dst, src);
#else
      *(float4*)dst = *(const float4*)src;
#endif
    }
#if HAVE_ASYNC_LDS
    async_wait0();  // issuing wave drains ASYNCcnt before the barrier
#endif
    if (tid < 32) {
      float s = sRedSum[0][tid];
      for (int w2 = 1; w2 < 16; ++w2) s += sRedSum[w2][tid];
      float lp = -__logf(s);
      if (sEmit[tid]) sScore[tid] += lp;
      out[(size_t)tid * 1000 + t] = (float)(sEmit[tid] ? sPos[tid] : 0);
    }
    __syncthreads();

    lstm_gemm();
    __syncthreads();
    lstm_apply();
    __syncthreads();
  }

  if (tid < 32) {
    out[32000 + tid] = sScore[tid];
    float e = __expf(sScore[tid]);
    for (int d = 1; d < 32; d <<= 1) e += __shfl_xor(e, d, 32);
    if (tid == 0) out[32032] = e * (1.f / 32.f);
  }
}

// ---------------------------------------------------------------------------
extern "C" void kernel_launch(void* const* d_in, const int* in_sizes, int n_in,
                              void* d_out, int out_size, void* d_ws,
                              size_t ws_size, hipStream_t stream) {
  (void)in_sizes; (void)n_in; (void)out_size; (void)ws_size;
  const float* enc = (const float*)d_in[0];
  const float* embed = (const float*)d_in[1];
  const float* W_ih = (const float*)d_in[2];
  const float* W_hh = (const float*)d_in[3];
  const float* b_lstm = (const float*)d_in[4];
  const float* W_enc = (const float*)d_in[5];
  const float* W_dec = (const float*)d_in[6];
  const float* b_joint = (const float*)d_in[7];
  const float* W_out = (const float*)d_in[8];
  float* out = (float*)d_out;

  char* ws = (char*)d_ws;  // requires ~106 MB workspace
  _Float16* encF16 = (_Float16*)(ws);               // 32,768,000 B
  float* encPrj = (float*)(ws + 32768000);          // 65,536,000 B
  float4* WencP = (float4*)(ws + 98304000);         //    524,288 B
  float4* WdecP = (float4*)(ws + 98828288);         //    524,288 B
  float4* WoutP = (float4*)(ws + 99352576);         //  1,048,576 B
  float4* WlstmP = (float4*)(ws + 100401152);       //  4,194,304 B
  _Float16* embF16 = (_Float16*)(ws + 104595456);   //  1,048,576 B

  k_f32_to_f16<<<4096, 256, 0, stream>>>(enc, encF16, 32 * 1000 * 512);
  k_f32_to_f16<<<512, 256, 0, stream>>>(embed, embF16, 1024 * 512);
  k_pack_b<<<128, 128, 0, stream>>>(W_enc, W_enc, 512, 512, 512, WencP);
  k_pack_b<<<128, 128, 0, stream>>>(W_dec, W_dec, 512, 512, 512, WdecP);
  k_pack_b<<<256, 128, 0, stream>>>(W_out, W_out, 512, 512, 1024, WoutP);
  k_pack_b<<<1024, 128, 0, stream>>>(W_ih, W_hh, 512, 1024, 2048, WlstmP);
  k_enc_proj<<<2000, 256, 0, stream>>>(encF16, WencP, encPrj);
  k_decode<<<1, 512, 0, stream>>>(encPrj, embF16, b_lstm, b_joint, WdecP, WoutP,
                                  WlstmP, out);
}
